// SimpleDocProcModel_76647986364631
// MI455X (gfx1250) — matrix operand order repaired and checked by reading, not compiled, not run on hardware
//
#include <hip/hip_runtime.h>
#include <hip/hip_bf16.h>
#include <math.h>

// ---------------------------------------------------------------------------
// SimpleDocProcModel on MI455X (gfx1250, wave32, WMMA).
//
// Live graph (ITERATIONS=1, only ll returned; Wh/bh/WD/bD are dead):
//   uu = vv@WA + bA                      [N,310]x[310,100]
//   ww = [uu, gather(uu,idx)*mask]      [N,500]  (hh==0 -> WB rows 500..599 dead)
//   bb = relu(ww@WB[0:500] + bB)         [N,500]x[500,100]
//   oo = tanh(bb@Wo + bo)                [N,100]x[100,100]
//   ll = oo@WC + bC                      [N,100]x[100,2]  (VALU reduce, 2 cols)
//
// ~24 GFLOP vs ~280MB traffic (gather fused into GEMM2 through LDS; no ww
// materialization) -> memory bound at 23.3 TB/s, ~12us floor.  f16 WMMA
// (v_wmma_f32_16x16x32_f16, f32 accum) keeps compute far below that.
// ---------------------------------------------------------------------------

typedef __attribute__((ext_vector_type(16))) _Float16 v16h;
typedef __attribute__((ext_vector_type(8)))  _Float16 v8h;
typedef __attribute__((ext_vector_type(8)))  float    v8f;

#define H100 100
#define LDU  112   // padded column stride for 100-wide activations (7 tiles of 16)
#define LDWW 512   // padded width of the gathered concat row (500 -> 16 k-tiles)
#define CT7  7     // 112/16 column tiles

__device__ __forceinline__ v8f wmma16(v16h a, v16h b, v8f c) {
  // D = A(16x32,f16) x B(32x16,f16) + C(16x16,f32)
  return __builtin_amdgcn_wmma_f32_16x16x32_f16(
      /*neg_a=*/false, a, /*neg_b=*/false, b,
      /*c_mod=*/(short)0, c, /*reuse_a=*/false, /*reuse_b=*/false);
}

// ---------------------------------------------------------------------------
// Pre-pack a row-major f32 weight matrix [K_eff x 100] (ld = ldw) into per-lane
// WMMA B-fragments (f16). One wave per (ktile, ctile) fragment.
// B(32x16) 16-bit layout: lanes 0-15 hold K=0..7,16..23 (col = lane), lanes 16-31
// hold K=8..15,24..31.  Fragment = 32 lanes x v16h = 1024 bytes.
// ---------------------------------------------------------------------------
__global__ __launch_bounds__(32) void prep_frags(const float* __restrict__ W,
                                                 int K_eff, int ldw,
                                                 _Float16* __restrict__ frags,
                                                 int ntiles) {
  int blk  = blockIdx.x;
  int kt   = blk / ntiles;
  int ct   = blk % ntiles;
  int lane = threadIdx.x & 31;
  int nl   = lane & 15;
  int grp  = lane >> 4;
  int n    = ct * 16 + nl;
  v16h f;
#pragma unroll
  for (int j = 0; j < 16; ++j) {
    int k   = kt * 32 + grp * 8 + (j & 7) + (j >> 3) * 16;
    float w = (k < K_eff && n < H100) ? W[(size_t)k * ldw + n] : 0.0f;
    f[j] = (_Float16)w;
  }
  ((v16h*)frags)[(size_t)blk * 32 + lane] = f;
}

// ---------------------------------------------------------------------------
// GEMM1: uu = vv(f32,[N x 310]) @ WA + bA  ->  uu f16 [N x LDU], pad cols = 0.
// A(16x32) f16 layout: lane = M (mod 16); halfs 0..7  = K 8*grp+0..7,
// halfs 8..15 = K 16+8*grp+0..7.  f32 loaded + converted in-register.
// ---------------------------------------------------------------------------
__global__ __launch_bounds__(256) void gemm1_kernel(const float* __restrict__ vv,
                                                    const _Float16* __restrict__ frags,
                                                    const float* __restrict__ bA,
                                                    _Float16* __restrict__ uu, int N) {
  const int KT = 10, KDIM = 310;
  int wave = threadIdx.x >> 5, lane = threadIdx.x & 31;
  int nl = lane & 15, grp = lane >> 4;
  int row0 = (blockIdx.x * 8 + wave) * 16;
  if (row0 >= N) return;
  const float* arow = vv + (size_t)(row0 + nl) * KDIM;

  v8f acc[CT7];
#pragma unroll
  for (int ct = 0; ct < CT7; ++ct)
    acc[ct] = (v8f){0.f, 0.f, 0.f, 0.f, 0.f, 0.f, 0.f, 0.f};

  for (int kt = 0; kt < KT; ++kt) {
    int kb = kt * 32 + grp * 8;
    if (kt + 1 < KT) __builtin_prefetch(arow + kb + 32, 0, 0);
    v16h a;
#pragma unroll
    for (int j = 0; j < 8; ++j) {
      int k0 = kb + j, k1 = kb + 16 + j;
      a[j]     = (_Float16)((k0 < KDIM) ? arow[k0] : 0.0f);
      a[8 + j] = (_Float16)((k1 < KDIM) ? arow[k1] : 0.0f);
    }
    const v16h* bf = (const v16h*)frags + (size_t)kt * CT7 * 32 + lane;
#pragma unroll
    for (int ct = 0; ct < CT7; ++ct)
      acc[ct] = wmma16(a, bf[(size_t)ct * 32], acc[ct]);
  }
  // D layout: VGPR r, lane L -> row = r + 8*grp, col = L%16
#pragma unroll
  for (int ct = 0; ct < CT7; ++ct) {
    int n = ct * 16 + nl;
    float bias = (n < H100) ? bA[n] : 0.0f;
#pragma unroll
    for (int r = 0; r < 8; ++r) {
      float v = (n < H100) ? (acc[ct][r] + bias) : 0.0f;  // exact 0 in pad cols
      uu[(size_t)(row0 + r + 8 * grp) * LDU + n] = (_Float16)v;
    }
  }
}

// ---------------------------------------------------------------------------
// GEMM2 with fused gather: for each block of 64 rows, the concat row
//   ww[row] = [uu[row][0:100], uu[idx0]|0, ..., uu[idx3]|0, 0-pad to 512]
// is gathered directly into LDS (uint granularity; segment boundaries are
// even so a uint never straddles segments), then 4 waves run the WMMA K-loop
// reading A-fragments from LDS.  16-byte chunks are XOR-swizzled by row so
// the 16 rows of a fragment read land on different bank groups.
// bb = relu(ww @ WBfrags + bB) -> f16 [N x LDU]
// ---------------------------------------------------------------------------
__global__ __launch_bounds__(128) void gemm2_fused_kernel(
    const _Float16* __restrict__ uu, const int* __restrict__ indices,
    const _Float16* __restrict__ frags, const float* __restrict__ bB,
    _Float16* __restrict__ bb, int N) {
  const int KT = 16;
  __shared__ _Float16 sww[64 * LDWW];              // 64 rows x 512 halfs = 64KB
  int t = threadIdx.x;                             // 0..127
  int rowblk = blockIdx.x * 64;
  if (rowblk >= N) return;

  // ---- gather phase: 2 uints per thread per row ----
  unsigned* swwu = (unsigned*)sww;
  for (int r = 0; r < 64; ++r) {
    int row = rowblk + r;
#pragma unroll
    for (int p = 0; p < 2; ++p) {
      int u = t + p * 128;                         // uint index 0..255 in row
      unsigned val = 0u;
      if (u < 250) {
        int h = 2 * u;
        int seg = h / 100;                         // 0 = self, 1..4 = neighbor
        int off = h - seg * 100;
        int src = (seg == 0) ? row : indices[(size_t)row * 4 + (seg - 1)];
        if (src >= 0)
          val = *(const unsigned*)(uu + (size_t)src * LDU + off);
      }
      int chunk = u >> 2;                          // 16B chunk index 0..63
      int phys  = chunk ^ (r & 15);                // bank swizzle
      swwu[r * (LDWW / 2) + phys * 4 + (u & 3)] = val;
    }
  }
  __syncthreads();

  // ---- WMMA phase: 4 waves x 16-row tiles ----
  int wave = t >> 5, lane = t & 31;
  int nl = lane & 15, grp = lane >> 4;
  int row0 = rowblk + wave * 16;
  int rl = wave * 16 + nl;                         // lane's row within block

  v8f acc[CT7];
#pragma unroll
  for (int ct = 0; ct < CT7; ++ct)
    acc[ct] = (v8f){0.f, 0.f, 0.f, 0.f, 0.f, 0.f, 0.f, 0.f};

  for (int kt = 0; kt < KT; ++kt) {
    int c0 = kt * 4 + grp;                         // chunk of halfs kb..kb+7
    int c1 = c0 + 2;                               // chunk of halfs kb+16..+23
    const _Float16* base = sww + (size_t)rl * LDWW;
    v8h lo = *(const v8h*)(base + (c0 ^ (rl & 15)) * 8);
    v8h hi = *(const v8h*)(base + (c1 ^ (rl & 15)) * 8);
    v16h a;
#pragma unroll
    for (int j = 0; j < 8; ++j) { a[j] = lo[j]; a[8 + j] = hi[j]; }
    const v16h* bf = (const v16h*)frags + (size_t)kt * CT7 * 32 + lane;
#pragma unroll
    for (int ct = 0; ct < CT7; ++ct)
      acc[ct] = wmma16(a, bf[(size_t)ct * 32], acc[ct]);
  }
#pragma unroll
  for (int ct = 0; ct < CT7; ++ct) {
    int n = ct * 16 + nl;
    float bias = (n < H100) ? bB[n] : 0.0f;
#pragma unroll
    for (int r = 0; r < 8; ++r) {
      float v = fmaxf(acc[ct][r] + bias, 0.0f);
      if (n >= H100) v = 0.0f;                     // keep pad cols exact 0
      bb[(size_t)(row0 + r + 8 * grp) * LDU + n] = (_Float16)v;
    }
  }
}

// ---------------------------------------------------------------------------
// GEMM3 fused: oo = tanh(bb @ Wofrags + bo); ll = oo @ WC + bC -> d_out [N,2]
// K padded to 128: k in [100,128) hits zeroed fragment rows, so the (finite)
// over-read garbage contributes exactly 0.  oo tile goes through LDS, then
// each lane reduces one (row, class) pair against WC (only 2 output cols).
// ---------------------------------------------------------------------------
__global__ __launch_bounds__(256) void gemm3_kernel(const _Float16* __restrict__ bb,
                                                    const _Float16* __restrict__ frags,
                                                    const float* __restrict__ bo,
                                                    const float* __restrict__ WC,
                                                    const float* __restrict__ bC,
                                                    float* __restrict__ out, int N) {
  const int KT = 4;
  __shared__ float lds[8][16][LDU];
  int wave = threadIdx.x >> 5, lane = threadIdx.x & 31;
  int nl = lane & 15, grp = lane >> 4;
  int row0 = (blockIdx.x * 8 + wave) * 16;
  if (row0 >= N) return;
  const _Float16* arow = bb + (size_t)(row0 + nl) * LDU;

  v8f acc[CT7];
#pragma unroll
  for (int ct = 0; ct < CT7; ++ct)
    acc[ct] = (v8f){0.f, 0.f, 0.f, 0.f, 0.f, 0.f, 0.f, 0.f};

  for (int kt = 0; kt < KT; ++kt) {
    int kb = kt * 32 + grp * 8;
    v8h lo = *(const v8h*)(arow + kb);        // may over-read into next row:
    v8h hi = *(const v8h*)(arow + kb + 16);   // finite data x zero weights = 0
    v16h a;
#pragma unroll
    for (int j = 0; j < 8; ++j) { a[j] = lo[j]; a[8 + j] = hi[j]; }
    const v16h* bf = (const v16h*)frags + (size_t)kt * CT7 * 32 + lane;
#pragma unroll
    for (int ct = 0; ct < CT7; ++ct)
      acc[ct] = wmma16(a, bf[(size_t)ct * 32], acc[ct]);
  }
#pragma unroll
  for (int ct = 0; ct < CT7; ++ct) {
    int n = ct * 16 + nl;
    float bias = (n < H100) ? bo[n] : 0.0f;
#pragma unroll
    for (int r = 0; r < 8; ++r) {
      float v = (n < H100) ? tanhf(acc[ct][r] + bias) : 0.0f;
      lds[wave][r + 8 * grp][n] = v;
    }
  }
  __syncthreads();
  // lane -> (row = lane%16, class c = lane/16); ll = oo @ WC + bC
  int rowl = nl, c = grp;
  float s = bC[c];
#pragma unroll 4
  for (int n = 0; n < H100; ++n)
    s += lds[wave][rowl][n] * WC[n * 2 + c];
  out[(size_t)(row0 + rowl) * 2 + c] = s;
}

// ---------------------------------------------------------------------------
// Host-side launch.  Inputs (setup_inputs order):
//  0 indices  1 vv  2 num_words  3 WA  4 bA  5 WB  6 bB  7 Wo  8 bo
//  9 Wh  10 bh  11 WC  12 bC  13 WD  14 bD
// ws layout: [fragsA 70KB][fragsB 112KB][fragsO 28KB][uu 29.4MB][bb 29.4MB]
// (bb cannot alias uu: the fused GEMM2 gathers arbitrary uu rows while
//  writing bb.)
// ---------------------------------------------------------------------------
extern "C" void kernel_launch(void* const* d_in, const int* in_sizes, int n_in,
                              void* d_out, int out_size, void* d_ws, size_t ws_size,
                              hipStream_t stream) {
  const int*   indices = (const int*)d_in[0];
  const float* vv = (const float*)d_in[1];
  const float* WA = (const float*)d_in[3];
  const float* bA = (const float*)d_in[4];
  const float* WB = (const float*)d_in[5];
  const float* bB = (const float*)d_in[6];
  const float* Wo = (const float*)d_in[7];
  const float* bo = (const float*)d_in[8];
  const float* WC = (const float*)d_in[11];
  const float* bC = (const float*)d_in[12];
  float* out = (float*)d_out;
  int N = in_sizes[1] / 310;

  char* ws = (char*)d_ws;
  _Float16* fragsA = (_Float16*)(ws);                       // 10x7 frags * 1KB
  _Float16* fragsB = (_Float16*)(ws + 71680);               // 16x7 frags
  _Float16* fragsO = (_Float16*)(ws + 71680 + 114688);      // 4x7 frags
  _Float16* uu     = (_Float16*)(ws + 215040);              // N*LDU f16
  _Float16* bbuf   = (_Float16*)(ws + 215040 + (size_t)N * LDU * 2);

  prep_frags<<<10 * CT7, 32, 0, stream>>>(WA, 310, H100, fragsA, CT7);
  prep_frags<<<16 * CT7, 32, 0, stream>>>(WB, 500, H100, fragsB, CT7);
  prep_frags<<< 4 * CT7, 32, 0, stream>>>(Wo, 100, H100, fragsO, CT7);

  int blocks128 = (N + 127) / 128;
  int blocks64  = (N + 63) / 64;
  gemm1_kernel<<<blocks128, 256, 0, stream>>>(vv, fragsA, bA, uu, N);
  gemm2_fused_kernel<<<blocks64, 128, 0, stream>>>(uu, indices, fragsB, bB, bbuf, N);
  gemm3_kernel<<<blocks128, 256, 0, stream>>>(bbuf, fragsO, bo, WC, bC, out, N);
}
